// GCNConvGNNLayer_85744727097461
// MI455X (gfx1250) — compile-verified, matrix operand order checked by reading, not run on hardware
//
#include <hip/hip_runtime.h>
#include <hip/hip_bf16.h>

typedef __attribute__((ext_vector_type(16))) _Float16 v16h;
typedef __attribute__((ext_vector_type(8)))  float    v8f;

#define N_NODES   100000
#define N_EDGES   1600000
#define DIM       128
#define N_FILTERS 64
#define N_GAUSS   50
#define CUTOFF    5.0f

// K index mapping for 16x32 f16 A/B fragments (ISA 7.12.2, wave32):
// lane 0-15: elements 0..7 -> K=0..7,  elements 8..15 -> K=16..23
// lane 16-31: elements 0..7 -> K=8..15, elements 8..15 -> K=24..31
__device__ __forceinline__ int frag_k(int lane, int j) {
    int kb = (lane >> 4) * 8;
    return (j < 8) ? (kb + j) : (16 + kb + (j - 8));
}

__global__ void zero_kernel(float* __restrict__ p, long n) {
    long i = (long)blockIdx.x * blockDim.x + threadIdx.x;
    long stride = (long)gridDim.x * blockDim.x;
    for (; i < n; i += stride) p[i] = 0.0f;
}

// h[N,64] = x[N,128] @ lin1_w[64,128]^T   (one wave per 16-node tile)
__global__ void lin1_kernel(const float* __restrict__ x,
                            const float* __restrict__ w,
                            float* __restrict__ h) {
    int wave = (int)((blockIdx.x * blockDim.x + threadIdx.x) >> 5);
    int lane = threadIdx.x & 31;
    if (wave * 16 >= N_NODES) return;
    int row0 = wave * 16;
    int m = lane & 15;
    int node = row0 + m;
    bool valid = node < N_NODES;

    v8f acc[4] = {};
    #pragma unroll
    for (int ks = 0; ks < 4; ++ks) {
        v16h a;
        #pragma unroll
        for (int j = 0; j < 16; ++j) {
            int k = ks * 32 + frag_k(lane, j);
            a[j] = valid ? (_Float16)x[(long)node * DIM + k] : (_Float16)0.0f;
        }
        #pragma unroll
        for (int nt = 0; nt < 4; ++nt) {
            int f = nt * 16 + m;               // B fragment: row = output filter
            v16h b;
            #pragma unroll
            for (int j = 0; j < 16; ++j) {
                int k = ks * 32 + frag_k(lane, j);
                b[j] = (_Float16)w[f * DIM + k];
            }
            acc[nt] = __builtin_amdgcn_wmma_f32_16x16x32_f16(
                false, a, false, b, (short)0, acc[nt], false, false);
        }
    }
    int rbase = (lane >> 4) * 8;               // D layout: element r -> M = rbase+r
    #pragma unroll
    for (int nt = 0; nt < 4; ++nt) {
        #pragma unroll
        for (int r = 0; r < 8; ++r) {
            int nn = row0 + rbase + r;
            if (nn < N_NODES) h[(long)nn * N_FILTERS + nt * 16 + m] = acc[nt][r];
        }
    }
}

// Per 16-edge tile: gaussian smear -> MLP (2 WMMA GEMMs) -> cutoff scale ->
// gather h[src] -> atomic scatter-add into agg[dst].
__global__ void __launch_bounds__(256) edge_kernel(
        const float* __restrict__ edge_w,      // dist [E]
        const int*   __restrict__ eidx,        // [2,E]
        const float* __restrict__ enn1_w,      // [64,50]
        const float* __restrict__ enn1_b,      // [64]
        const float* __restrict__ enn2_w,      // [64,64]
        const float* __restrict__ enn2_b,      // [64]
        const float* __restrict__ h,           // [N,64]
        float*       __restrict__ agg) {       // [N,64]
    __shared__ float lds[8][16][N_FILTERS + 1];  // +1 pad: conflict-free transpose
    int wid  = threadIdx.x >> 5;
    int lane = threadIdx.x & 31;
    long tile = (long)blockIdx.x * 8 + wid;
    long e0 = tile * 16;
    if (e0 >= N_EDGES) return;                 // never taken: 12500*8*16 == E
    int m = lane & 15;
    int rbase = (lane >> 4) * 8;

    const float delta = CUTOFF / (float)(N_GAUSS - 1);
    const float coeff = -0.5f / (delta * delta);
    float d = edge_w[e0 + m];                  // this lane's edge distance (A rows)

    // ---- GEMM1: smeared[16 x 64(pad)] @ enn1^T -> h1[16 x 64]
    v8f acc1[4] = {};
    #pragma unroll
    for (int ks = 0; ks < 2; ++ks) {
        v16h a;
        #pragma unroll
        for (int j = 0; j < 16; ++j) {
            int k = ks * 32 + frag_k(lane, j);
            float v = 0.0f;
            if (k < N_GAUSS) {
                float t = d - (float)k * delta;
                v = __expf(coeff * t * t);
            }
            a[j] = (_Float16)v;
        }
        #pragma unroll
        for (int nt = 0; nt < 4; ++nt) {
            int f = nt * 16 + m;
            v16h b;
            #pragma unroll
            for (int j = 0; j < 16; ++j) {
                int k = ks * 32 + frag_k(lane, j);
                b[j] = (k < N_GAUSS) ? (_Float16)enn1_w[f * N_GAUSS + k]
                                     : (_Float16)0.0f;
            }
            acc1[nt] = __builtin_amdgcn_wmma_f32_16x16x32_f16(
                false, a, false, b, (short)0, acc1[nt], false, false);
        }
    }
    // bias + relu; D-layout -> LDS [edge_row][filter] (transpose for next A frag)
    #pragma unroll
    for (int nt = 0; nt < 4; ++nt) {
        float bias = enn1_b[nt * 16 + m];
        #pragma unroll
        for (int r = 0; r < 8; ++r) {
            float v = acc1[nt][r] + bias;
            lds[wid][rbase + r][nt * 16 + m] = v > 0.0f ? v : 0.0f;
        }
    }
    // intra-wave cross-lane LDS RAW: wait own DS stores (CDNA5 split counter)
    asm volatile("s_wait_dscnt 0" ::: "memory");

    // overlap: fetch edge meta for scatter phase while GEMM2 runs
    const float PI = 3.14159265358979323846f;
    float Cv[8]; int se[8]; int de[8];
    #pragma unroll
    for (int r = 0; r < 8; ++r) {
        long e = e0 + rbase + r;
        float dd = edge_w[e];
        Cv[r] = 0.5f * (__cosf(dd * (PI / CUTOFF)) + 1.0f);
        se[r] = eidx[e];
        de[r] = eidx[(long)N_EDGES + e];
    }

    // ---- GEMM2: h1[16x64] @ enn2^T -> W[16x64]
    v8f acc2[4] = {};
    #pragma unroll
    for (int ks = 0; ks < 2; ++ks) {
        v16h a;
        #pragma unroll
        for (int j = 0; j < 16; ++j) {
            int k = ks * 32 + frag_k(lane, j);
            a[j] = (_Float16)lds[wid][m][k];
        }
        #pragma unroll
        for (int nt = 0; nt < 4; ++nt) {
            int f = nt * 16 + m;
            v16h b;
            #pragma unroll
            for (int j = 0; j < 16; ++j) {
                int k = ks * 32 + frag_k(lane, j);
                b[j] = (_Float16)enn2_w[f * N_FILTERS + k];
            }
            acc2[nt] = __builtin_amdgcn_wmma_f32_16x16x32_f16(
                false, a, false, b, (short)0, acc2[nt], false, false);
        }
    }

    // ---- cutoff scale, gather h[src], scatter-add agg[dst] (L2-resident)
    #pragma unroll
    for (int nt = 0; nt < 4; ++nt) {
        int n = nt * 16 + m;
        float bias = enn2_b[n];
        #pragma unroll
        for (int r = 0; r < 8; ++r) {
            float Wv = (acc2[nt][r] + bias) * Cv[r];
            float msg = Wv * h[(long)se[r] * N_FILTERS + n];
            atomicAdd(&agg[(long)de[r] * N_FILTERS + n], msg);
        }
    }
}

// out[N,128] = x + relu(agg[N,64] @ lin2_w[128,64]^T + b)
__global__ void out_kernel(const float* __restrict__ agg,
                           const float* __restrict__ w,
                           const float* __restrict__ bvec,
                           const float* __restrict__ x,
                           float* __restrict__ out) {
    int wave = (int)((blockIdx.x * blockDim.x + threadIdx.x) >> 5);
    int lane = threadIdx.x & 31;
    if (wave * 16 >= N_NODES) return;
    int row0 = wave * 16;
    int m = lane & 15;
    int node = row0 + m;
    bool valid = node < N_NODES;

    v8f acc[8] = {};
    #pragma unroll
    for (int ks = 0; ks < 2; ++ks) {
        v16h a;
        #pragma unroll
        for (int j = 0; j < 16; ++j) {
            int k = ks * 32 + frag_k(lane, j);
            a[j] = valid ? (_Float16)agg[(long)node * N_FILTERS + k] : (_Float16)0.0f;
        }
        #pragma unroll
        for (int nt = 0; nt < 8; ++nt) {
            int dcol = nt * 16 + m;
            v16h b;
            #pragma unroll
            for (int j = 0; j < 16; ++j) {
                int k = ks * 32 + frag_k(lane, j);
                b[j] = (_Float16)w[dcol * N_FILTERS + k];
            }
            acc[nt] = __builtin_amdgcn_wmma_f32_16x16x32_f16(
                false, a, false, b, (short)0, acc[nt], false, false);
        }
    }
    int rbase = (lane >> 4) * 8;
    #pragma unroll
    for (int nt = 0; nt < 8; ++nt) {
        int n = nt * 16 + m;
        float bias = bvec[n];
        #pragma unroll
        for (int r = 0; r < 8; ++r) {
            int nn = row0 + rbase + r;
            if (nn < N_NODES) {
                float y = acc[nt][r] + bias;
                y = y > 0.0f ? y : 0.0f;
                out[(long)nn * DIM + n] = x[(long)nn * DIM + n] + y;
            }
        }
    }
}

extern "C" void kernel_launch(void* const* d_in, const int* in_sizes, int n_in,
                              void* d_out, int out_size, void* d_ws, size_t ws_size,
                              hipStream_t stream) {
    const float* x           = (const float*)d_in[0];
    const int*   eidx        = (const int*)  d_in[1];
    const float* edge_weight = (const float*)d_in[2];
    // d_in[3] edge_attr == edge_weight (same dist); use edge_weight throughout
    const float* lin1_w      = (const float*)d_in[4];
    const float* lin2_w      = (const float*)d_in[5];
    const float* lin2_b      = (const float*)d_in[6];
    const float* enn1_w      = (const float*)d_in[7];
    const float* enn1_b      = (const float*)d_in[8];
    const float* enn2_w      = (const float*)d_in[9];
    const float* enn2_b      = (const float*)d_in[10];
    float* out = (float*)d_out;

    float* h   = (float*)d_ws;                           // [N,64]  25.6 MB
    float* agg = h + (size_t)N_NODES * N_FILTERS;        // [N,64]  25.6 MB

    zero_kernel<<<1024, 256, 0, stream>>>(agg, (long)N_NODES * N_FILTERS);

    int node_tiles = (N_NODES + 15) / 16;                // 6250
    int blocks_n   = (node_tiles + 7) / 8;               // 782 (8 waves/block)
    lin1_kernel<<<blocks_n, 256, 0, stream>>>(x, lin1_w, h);

    int edge_blocks = (N_EDGES / 16) / 8;                // 12500, exact
    edge_kernel<<<edge_blocks, 256, 0, stream>>>(edge_weight, eidx,
        enn1_w, enn1_b, enn2_w, enn2_b, h, agg);

    out_kernel<<<blocks_n, 256, 0, stream>>>(agg, lin2_w, lin2_b, x, out);
}
